// InterpoAttention_61125974557072
// MI455X (gfx1250) — matrix operand order repaired and checked by reading, not compiled, not run on hardware
//
#include <hip/hip_runtime.h>
#include <hip/hip_bf16.h>

#define B_  4
#define L_  1024
#define D_  1024
#define H_  16
#define T_  64
#define DH  64   // head dim

typedef __attribute__((ext_vector_type(16))) __bf16 v16bf;
typedef __attribute__((ext_vector_type(8)))  __bf16 v8bf;
typedef __attribute__((ext_vector_type(8)))  float  v8f;

union AFrag { v16bf v; v8bf h[2]; };

// float -> bf16, round-to-nearest-even
static __device__ __forceinline__ __bf16 f2bf(float f) {
  union { float f; unsigned u; } uf; uf.f = f;
  unsigned r = uf.u + 0x7FFFu + ((uf.u >> 16) & 1u);
  union { unsigned short s; __bf16 b; } ub;
  ub.s = (unsigned short)(r >> 16);
  return ub.b;
}

static __device__ __forceinline__ v8f vzero8() {
  v8f z;
#pragma unroll
  for (int i = 0; i < 8; ++i) z[i] = 0.0f;
  return z;
}

static __device__ __forceinline__ v8f wmma_bf16(v16bf a, v16bf b, v8f c) {
  return __builtin_amdgcn_wmma_f32_16x16x32_bf16(false, a, false, b, (short)0, c,
                                                 false, false);
}

// Async-DMA one contiguous 4KB (32x64 bf16) tile global->LDS, split over 4 waves.
// Each wave issues 2 x (32 lanes x 16B) = 1KB. Tracked by ASYNCcnt.
static __device__ __forceinline__ void async_copy_tile(const __bf16* gsrc,
                                                       __bf16* ldst,
                                                       int wid, int lane) {
  unsigned long long g = (unsigned long long)(uintptr_t)gsrc;
  unsigned l = (unsigned)(uintptr_t)ldst;
#pragma unroll
  for (int c = 0; c < 2; ++c) {
    unsigned off = (unsigned)(((wid * 2 + c) * 32 + lane) * 16);
    asm volatile("global_load_async_to_lds_b128 %0, %1, off"
                 :
                 : "v"(l + off), "v"(g + (unsigned long long)off)
                 : "memory");
  }
}

// ---------------------------------------------------------------- prep kernels

__global__ void cvt_f32_bf16_kernel(const float* __restrict__ src,
                                    __bf16* __restrict__ dst, int n) {
  for (int i = blockIdx.x * blockDim.x + threadIdx.x; i < n;
       i += gridDim.x * blockDim.x)
    dst[i] = f2bf(src[i]);
}

// Pb[b,h,i,dd] = bf16(x[b,i,h*64+dd])
__global__ void pack_p_kernel(const float* __restrict__ x,
                              __bf16* __restrict__ Pb) {
  const int n = B_ * L_ * D_;
  for (int idx = blockIdx.x * blockDim.x + threadIdx.x; idx < n;
       idx += gridDim.x * blockDim.x) {
    int col = idx & (D_ - 1);
    int row = idx >> 10;
    int h = col >> 6, dd = col & 63;
    int b = row >> 10, i = row & (L_ - 1);
    Pb[(((b * H_ + h) * L_ + i) << 6) + dd] = f2bf(x[idx]);
  }
}

__global__ void colsum_kernel(const float* __restrict__ mat,
                              float* __restrict__ out) {
  int l = blockIdx.x * blockDim.x + threadIdx.x;
  if (l >= L_) return;
  float s = 0.0f;
  for (int dIdx = 0; dIdx < D_; ++dIdx) s += mat[dIdx * L_ + l];
  out[l] = s;
}

__global__ void coords_kernel(const float* __restrict__ t_enc,
                              const float* __restrict__ Wt,
                              const float* __restrict__ bt,
                              const float* __restrict__ psum,
                              float* __restrict__ coords,
                              float* __restrict__ qsca) {
  int idx = blockIdx.x * blockDim.x + threadIdx.x;
  if (idx >= B_ * L_) return;
  float z = bt[0];
  const float* te = t_enc + idx * T_;
#pragma unroll 8
  for (int k = 0; k < T_; ++k) z += te[k] * Wt[k];
  float tau = 1.0f / (1.0f + __expf(-z));
  float c = tau * (float)(L_ - 1);
  coords[idx] = c;
  int i = (int)floorf(c);
  float t = c - (float)i;
  int im1 = min(max(i - 1, 0), L_ - 1);
  int ic0 = min(max(i, 0), L_ - 1);
  int ic1 = min(i + 1, L_ - 1);
  int ic2 = min(i + 2, L_ - 1);
  float p0 = psum[im1], p1 = psum[ic0], p2 = psum[ic1], p3 = psum[ic2];
  float t2 = t * t, t3 = t2 * t;
  qsca[idx] = 0.5f * (2.0f * p1 + (-p0 + p2) * t +
                      (2.0f * p0 - 5.0f * p1 + 4.0f * p2 - p3) * t2 +
                      (-p0 + 3.0f * p1 - 3.0f * p2 + p3) * t3);
}

// ------------------------------------------------------------- WMMA GEMM core
// Load one K-chunk's operands: A (16x32) for this wave's rows, B (4 tiles).
static __device__ __forceinline__ void load_frag(const __bf16* __restrict__ xrow,
                                                 const __bf16* __restrict__ Wb,
                                                 int n0, int ln, int hi, int kc,
                                                 AFrag& a, v16bf (&b)[4]) {
  const __bf16* ap = xrow + kc + 16 * hi;
  a.h[0] = *reinterpret_cast<const v8bf*>(ap);
  a.h[1] = *reinterpret_cast<const v8bf*>(ap + 8);
#pragma unroll
  for (int t = 0; t < 4; ++t)
    b[t] = *reinterpret_cast<const v16bf*>(Wb + (n0 + t * 16 + ln) * D_ + kc + 16 * hi);
}

// C[i,n] = sum_k xb[i,k]*Wb[n,k]; wave tile 16x64, K chunks of 32, double-buffered.
// vt_mode 0: bf16 out in [B,H,L,64]; vt_mode 1: bf16 out in [B,H,64,L].
__global__ void __launch_bounds__(128)
qkv_gemm_kernel(const __bf16* __restrict__ xb, const __bf16* __restrict__ Wb,
                __bf16* __restrict__ outp, int vt_mode) {
  int lane = threadIdx.x & 31;
  int wid  = threadIdx.x >> 5;
  int tile = blockIdx.x * 4 + wid;
  int mtile = tile >> 4;
  int ntile = tile & 15;
  int i0 = mtile * 16, n0 = ntile * 64;
  int ln = lane & 15, hi = lane >> 4;
  const __bf16* xrow = xb + (i0 + ln) * D_;

  v8f acc[4];
#pragma unroll
  for (int t = 0; t < 4; ++t) acc[t] = vzero8();

  AFrag a[2];
  v16bf bv[2][4];
  load_frag(xrow, Wb, n0, ln, hi, 0, a[0], bv[0]);
#pragma unroll 2
  for (int kc = 0; kc < D_; kc += 32) {
    int cur = (kc >> 5) & 1, nxt = cur ^ 1;
    if (kc + 32 < D_)
      load_frag(xrow, Wb, n0, ln, hi, kc + 32, a[nxt], bv[nxt]);
#pragma unroll
    for (int t = 0; t < 4; ++t)
      acc[t] = wmma_bf16(a[cur].v, bv[cur][t], acc[t]);
  }

#pragma unroll
  for (int v = 0; v < 8; ++v) {
    int m = i0 + v + 8 * hi;
    int b = m >> 10, il = m & (L_ - 1);
#pragma unroll
    for (int t = 0; t < 4; ++t) {
      int n = n0 + t * 16 + ln;
      int h = n >> 6, dd = n & 63;
      __bf16 val = f2bf(acc[t][v]);
      if (vt_mode == 0)
        outp[(((b * H_ + h) * L_ + il) << 6) + dd] = val;
      else
        outp[(((b * H_ + h) * DH + dd) << 10) + il] = val;
    }
  }
}

__global__ void __launch_bounds__(128)
out_gemm_kernel(const __bf16* __restrict__ ab, const __bf16* __restrict__ Wb,
                const float* __restrict__ bO, float* __restrict__ outp) {
  int lane = threadIdx.x & 31;
  int wid  = threadIdx.x >> 5;
  int tile = blockIdx.x * 4 + wid;
  int mtile = tile >> 4;
  int ntile = tile & 15;
  int i0 = mtile * 16, n0 = ntile * 64;
  int ln = lane & 15, hi = lane >> 4;
  const __bf16* arow = ab + (i0 + ln) * D_;

  v8f acc[4];
#pragma unroll
  for (int t = 0; t < 4; ++t) acc[t] = vzero8();

  AFrag a[2];
  v16bf bv[2][4];
  load_frag(arow, Wb, n0, ln, hi, 0, a[0], bv[0]);
#pragma unroll 2
  for (int kc = 0; kc < D_; kc += 32) {
    int cur = (kc >> 5) & 1, nxt = cur ^ 1;
    if (kc + 32 < D_)
      load_frag(arow, Wb, n0, ln, hi, kc + 32, a[nxt], bv[nxt]);
#pragma unroll
    for (int t = 0; t < 4; ++t)
      acc[t] = wmma_bf16(a[cur].v, bv[cur][t], acc[t]);
  }

#pragma unroll
  for (int v = 0; v < 8; ++v) {
    int m = i0 + v + 8 * hi;
#pragma unroll
    for (int t = 0; t < 4; ++t) {
      int n = n0 + t * 16 + ln;
      outp[m * D_ + n] = acc[t][v] + bO[n];
    }
  }
}

// ----------------------------------------------------- fused flash attention
// Block = 4 waves = 4 consecutive 16-query tiles of the SAME (b,h): the K/P
// j-tiles are shared, staged once per block via async-DMA to LDS (ping-pong).
__global__ void __launch_bounds__(128)
attn_kernel(const __bf16* __restrict__ Qb, const __bf16* __restrict__ Kb,
            const __bf16* __restrict__ Pb, const __bf16* __restrict__ Vt,
            const float* __restrict__ coords, const float* __restrict__ qsca,
            const float* __restrict__ rsum,
            const float* __restrict__ alpha, const float* __restrict__ beta,
            const float* __restrict__ gamma,
            __bf16* __restrict__ attn_out) {
  __shared__ float s_rsum[L_];
  __shared__ __align__(32) __bf16 s_probs[4 * 512];
  __shared__ __align__(16) __bf16 s_kt[2][32 * DH];
  __shared__ __align__(16) __bf16 s_pt[2][32 * DH];

  int lane = threadIdx.x & 31;
  int wid  = threadIdx.x >> 5;
  int w = blockIdx.x * 4 + wid;             // 4096 waves total
  int itile = w & 63;
  int h = (w >> 6) & (H_ - 1);
  int b = w >> 10;
  int i0 = itile * 16;
  int ln = lane & 15, hi = lane >> 4;

  const __bf16* Qbh = Qb + (size_t)(b * H_ + h) * L_ * DH;
  const __bf16* Kbh = Kb + (size_t)(b * H_ + h) * L_ * DH;
  const __bf16* Pbh = Pb + (size_t)(b * H_ + h) * L_ * DH;
  const __bf16* Vth = Vt + (size_t)(b * H_ + h) * DH * L_;

  // kick off DMA of the first K/P tile pair while we set up
  async_copy_tile(Kbh, &s_kt[0][0], wid, lane);
  async_copy_tile(Pbh, &s_pt[0][0], wid, lane);

  for (int t = threadIdx.x; t < L_; t += blockDim.x) s_rsum[t] = rsum[t];

  int mrow = i0 + ln;
  AFrag a_q[2], a_p[2];
#pragma unroll
  for (int kc2 = 0; kc2 < 2; ++kc2) {
    const __bf16* qp = Qbh + mrow * DH + kc2 * 32 + 16 * hi;
    a_q[kc2].h[0] = *reinterpret_cast<const v8bf*>(qp);
    a_q[kc2].h[1] = *reinterpret_cast<const v8bf*>(qp + 8);
    const __bf16* pp = Pbh + mrow * DH + kc2 * 32 + 16 * hi;
    a_p[kc2].h[0] = *reinterpret_cast<const v8bf*>(pp);
    a_p[kc2].h[1] = *reinterpret_cast<const v8bf*>(pp + 8);
  }

  float ci[8], qi[8];
#pragma unroll
  for (int v = 0; v < 8; ++v) {
    int r = i0 + v + 8 * hi;
    ci[v] = coords[b * L_ + r];
    qi[v] = qsca[b * L_ + r];
  }
  float al = alpha[h], be = beta[h], ga = gamma[h];

  float mstat[8], lstat[8];
#pragma unroll
  for (int v = 0; v < 8; ++v) { mstat[v] = -1e30f; lstat[v] = 0.0f; }
  v8f acc[4];
#pragma unroll
  for (int t = 0; t < 4; ++t) acc[t] = vzero8();

  __bf16* probs_base = s_probs + wid * 512;

  for (int j0 = 0; j0 < L_; j0 += 32) {
    int buf = (j0 >> 5) & 1;
    // prefetch next j-chunk's K/P tiles into the other LDS buffer, then wait
    // only for the CURRENT buffer's 4 outstanding DMAs (in-order completion).
    if (j0 + 32 < L_) {
      async_copy_tile(Kbh + (size_t)(j0 + 32) * DH, &s_kt[buf ^ 1][0], wid, lane);
      async_copy_tile(Pbh + (size_t)(j0 + 32) * DH, &s_pt[buf ^ 1][0], wid, lane);
      asm volatile("s_wait_asynccnt 0x4" ::: "memory");
    } else {
      asm volatile("s_wait_asynccnt 0x0" ::: "memory");
    }
    __syncthreads();   // (A) all waves' DMA into current buffer visible

    v8f sc[2], pij[2];
    sc[0] = sc[1] = pij[0] = pij[1] = vzero8();
#pragma unroll
    for (int kc2 = 0; kc2 < 2; ++kc2) {
#pragma unroll
      for (int t = 0; t < 2; ++t) {
        int jl = t * 16 + ln;
        v16bf bk = *reinterpret_cast<const v16bf*>(&s_kt[buf][jl * DH + kc2 * 32 + 16 * hi]);
        sc[t] = wmma_bf16(a_q[kc2].v, bk, sc[t]);
        v16bf bp = *reinterpret_cast<const v16bf*>(&s_pt[buf][jl * DH + kc2 * 32 + 16 * hi]);
        pij[t] = wmma_bf16(a_p[kc2].v, bp, pij[t]);
      }
    }
    if (j0 + 32 < L_) {  // keep the V stream warm in L2/L0
#pragma unroll
      for (int dt = 0; dt < 4; ++dt)
        __builtin_prefetch(Vth + (size_t)(dt * 16 + ln) * L_ + j0 + 32, 0, 1);
    }

    float cj[2], qj[2];
#pragma unroll
    for (int t = 0; t < 2; ++t) {
      int j = j0 + t * 16 + ln;
      cj[t] = coords[b * L_ + j];
      qj[t] = qsca[b * L_ + j];
    }

    float rowmax[8];
#pragma unroll
    for (int v = 0; v < 8; ++v) {
#pragma unroll
      for (int t = 0; t < 2; ++t) {
        float delta = fabsf(ci[v] - cj[t]);
        int ii = (int)delta;
        float tt = delta - (float)ii;
        int iA = min(ii, L_ - 1);
        int iB = min(ii + 1, L_ - 1);
        float rv = s_rsum[iA] * (1.0f - tt) + s_rsum[iB] * tt;
        float S = al * (qi[v] + qj[t]) + be * rv + ga;
        sc[t][v] = sc[t][v] * 0.125f + S * pij[t][v];   // 1/sqrt(64)
      }
      rowmax[v] = fmaxf(sc[0][v], sc[1][v]);
    }
#pragma unroll
    for (int mask = 1; mask <= 8; mask <<= 1)
#pragma unroll
      for (int v = 0; v < 8; ++v)
        rowmax[v] = fmaxf(rowmax[v], __shfl_xor(rowmax[v], mask, 32));

    float scalef[8], rowsum[8];
#pragma unroll
    for (int v = 0; v < 8; ++v) {
      float mnew = fmaxf(mstat[v], rowmax[v]);
      scalef[v] = __expf(mstat[v] - mnew);
      mstat[v] = mnew;
      float p0 = __expf(sc[0][v] - mnew);
      float p1 = __expf(sc[1][v] - mnew);
      sc[0][v] = p0; sc[1][v] = p1;
      rowsum[v] = p0 + p1;
    }
#pragma unroll
    for (int mask = 1; mask <= 8; mask <<= 1)
#pragma unroll
      for (int v = 0; v < 8; ++v)
        rowsum[v] += __shfl_xor(rowsum[v], mask, 32);
#pragma unroll
    for (int v = 0; v < 8; ++v) lstat[v] = lstat[v] * scalef[v] + rowsum[v];
#pragma unroll
    for (int dt = 0; dt < 4; ++dt)
#pragma unroll
      for (int v = 0; v < 8; ++v) acc[dt][v] *= scalef[v];

    // probs (D layout) -> LDS 16x32 tile -> reload as WMMA A operand
#pragma unroll
    for (int v = 0; v < 8; ++v) {
      int r = v + 8 * hi;
      probs_base[r * 32 + ln]      = f2bf(sc[0][v]);
      probs_base[r * 32 + 16 + ln] = f2bf(sc[1][v]);
    }
    __syncthreads();   // (B) probs ready; also fences K/P reads before rebuffer
    AFrag a_pr;
    const __bf16* prp = probs_base + ln * 32 + 16 * hi;
    a_pr.h[0] = *reinterpret_cast<const v8bf*>(prp);
    a_pr.h[1] = *reinterpret_cast<const v8bf*>(prp + 8);
#pragma unroll
    for (int dt = 0; dt < 4; ++dt) {
      v16bf bv = *reinterpret_cast<const v16bf*>(Vth + (size_t)(dt * 16 + ln) * L_ + j0 + 16 * hi);
      acc[dt] = wmma_bf16(a_pr.v, bv, acc[dt]);
    }
  }

#pragma unroll
  for (int v = 0; v < 8; ++v) {
    float inv = 1.0f / lstat[v];
    int r = i0 + v + 8 * hi;
#pragma unroll
    for (int dt = 0; dt < 4; ++dt) {
      int col = h * DH + dt * 16 + ln;
      attn_out[(size_t)(b * L_ + r) * D_ + col] = f2bf(acc[dt][v] * inv);
    }
  }
}

// ------------------------------------------------------------------- launcher

extern "C" void kernel_launch(void* const* d_in, const int* in_sizes, int n_in,
                              void* d_out, int out_size, void* d_ws, size_t ws_size,
                              hipStream_t stream) {
  const float* x     = (const float*)d_in[0];
  const float* t_enc = (const float*)d_in[1];
  const float* WQ    = (const float*)d_in[2];
  const float* WK    = (const float*)d_in[3];
  const float* WV    = (const float*)d_in[4];
  const float* WO    = (const float*)d_in[5];
  const float* bO    = (const float*)d_in[6];
  const float* Wt    = (const float*)d_in[7];
  const float* bt    = (const float*)d_in[8];
  const float* p_mat = (const float*)d_in[9];
  const float* r_mat = (const float*)d_in[10];
  const float* alpha = (const float*)d_in[11];
  const float* beta  = (const float*)d_in[12];
  const float* gamma = (const float*)d_in[13];
  float* out = (float*)d_out;

  char* ws = (char*)d_ws;
  auto take = [&](size_t bytes) {
    char* p = ws;
    ws += (bytes + 255) & ~(size_t)255;
    return p;
  };
  __bf16* xb     = (__bf16*)take((size_t)B_ * L_ * D_ * 2);
  __bf16* WQb    = (__bf16*)take((size_t)D_ * D_ * 2);
  __bf16* WKb    = (__bf16*)take((size_t)D_ * D_ * 2);
  __bf16* WVb    = (__bf16*)take((size_t)D_ * D_ * 2);
  __bf16* WOb    = (__bf16*)take((size_t)D_ * D_ * 2);
  __bf16* Qb     = (__bf16*)take((size_t)B_ * L_ * D_ * 2);
  __bf16* Kb     = (__bf16*)take((size_t)B_ * L_ * D_ * 2);
  __bf16* Pb     = (__bf16*)take((size_t)B_ * L_ * D_ * 2);
  __bf16* Vt     = (__bf16*)take((size_t)B_ * L_ * D_ * 2);
  __bf16* aout   = (__bf16*)take((size_t)B_ * L_ * D_ * 2);
  float*  coords = (float*)take((size_t)B_ * L_ * 4);
  float*  qsca   = (float*)take((size_t)B_ * L_ * 4);
  float*  psum   = (float*)take((size_t)L_ * 4);
  float*  rsum   = (float*)take((size_t)L_ * 4);

  cvt_f32_bf16_kernel<<<1024, 256, 0, stream>>>(x, xb, B_ * L_ * D_);
  cvt_f32_bf16_kernel<<<512, 256, 0, stream>>>(WQ, WQb, D_ * D_);
  cvt_f32_bf16_kernel<<<512, 256, 0, stream>>>(WK, WKb, D_ * D_);
  cvt_f32_bf16_kernel<<<512, 256, 0, stream>>>(WV, WVb, D_ * D_);
  cvt_f32_bf16_kernel<<<512, 256, 0, stream>>>(WO, WOb, D_ * D_);
  pack_p_kernel<<<1024, 256, 0, stream>>>(x, Pb);

  colsum_kernel<<<(L_ + 255) / 256, 256, 0, stream>>>(p_mat, psum);
  colsum_kernel<<<(L_ + 255) / 256, 256, 0, stream>>>(r_mat, rsum);
  coords_kernel<<<(B_ * L_ + 255) / 256, 256, 0, stream>>>(t_enc, Wt, bt, psum,
                                                           coords, qsca);

  qkv_gemm_kernel<<<1024, 128, 0, stream>>>(xb, WQb, Qb, 0);
  qkv_gemm_kernel<<<1024, 128, 0, stream>>>(xb, WKb, Kb, 0);
  qkv_gemm_kernel<<<1024, 128, 0, stream>>>(xb, WVb, Vt, 1);

  attn_kernel<<<1024, 128, 0, stream>>>(Qb, Kb, Pb, Vt, coords, qsca, rsum,
                                        alpha, beta, gamma, aout);

  out_gemm_kernel<<<1024, 128, 0, stream>>>(aout, WOb, bO, out);
}
